// RNNLayer_42047729828016
// MI455X (gfx1250) — matrix operand order repaired and checked by reading, not compiled, not run on hardware
//
#include <hip/hip_runtime.h>
#include <hip/hip_bf16.h>

// ---------------------------------------------------------------------------
// Types for WMMA fragments (CDNA5 / gfx1250, wave32)
// ---------------------------------------------------------------------------
typedef __attribute__((ext_vector_type(16))) __bf16    v16bf;
typedef __attribute__((ext_vector_type(8)))  float     v8f;
typedef __attribute__((ext_vector_type(4)))  unsigned  u32x4;

union Frag16 {           // 16 bf16 = 32 bytes = two 16B chunks
    u32x4 u[2];
    v16bf v;
};

__device__ __forceinline__ unsigned short f32_to_bf16_rne(float f) {
    unsigned u = __float_as_uint(f);
    unsigned r = (u + 0x7FFFu + ((u >> 16) & 1u)) >> 16;   // round-nearest-even
    return (unsigned short)r;
}

__device__ __forceinline__ u32x4 load16B(const unsigned short* p) {
    return *reinterpret_cast<const u32x4*>(p);
}

// async copy: 16B global -> LDS, tracked by ASYNCcnt (CDNA5 async-LDS path)
__device__ __forceinline__ void asyncCopy16B(unsigned ldsAddr,
                                             const unsigned short* gAddr) {
    asm volatile("global_load_async_to_lds_b128 %0, %1, off"
                 :: "v"(ldsAddr), "v"(gAddr) : "memory");
}
__device__ __forceinline__ void waitAsync0() {
    asm volatile("s_wait_asynccnt 0x0" ::: "memory");
}

// ---------------------------------------------------------------------------
// 1) fp32 -> bf16 elementwise convert (weights)
// ---------------------------------------------------------------------------
__global__ void k_convert_bf16(const float* __restrict__ src,
                               unsigned short* __restrict__ dst, int n) {
    int i = blockIdx.x * blockDim.x + threadIdx.x;
    if (i < n) dst[i] = f32_to_bf16_rne(src[i]);
}

// ---------------------------------------------------------------------------
// 2) embedding gather + convert:  x_bf16[bs, h] = bf16(emb[ids[bs], h])
// ---------------------------------------------------------------------------
__global__ void k_gather_emb(const int* __restrict__ ids,
                             const float* __restrict__ emb,
                             unsigned short* __restrict__ x,
                             int H) {
    long long i = (long long)blockIdx.x * blockDim.x + threadIdx.x; // over BS*H
    int row = (int)(i >> 10);      // H = 1024
    int col = (int)(i & 1023);
    long long src = (long long)ids[row] * H + col;
    x[i] = f32_to_bf16_rne(emb[src]);
}

// ---------------------------------------------------------------------------
// 3) bf16 WMMA GEMM:  C[M,N] (f32) = A[M,K] * W[N,K]^T
//    Block = 256 threads (8 waves), tile = 128(M) x 128(N).
//    B tile (128x32 bf16 = 8 KB) double-buffered in LDS, staged with
//    global_load_async_to_lds_b128 overlapping the 8 WMMAs of each K-step;
//    s_wait_asynccnt 0 + barrier publish the back buffer.  Final K-step is
//    peeled so the steady-state loop has no guards/selects.
// ---------------------------------------------------------------------------
__global__ void __launch_bounds__(256)
k_wmma_gemm_bf16(const unsigned short* __restrict__ A,   // [M x K] bf16 row-major
                 const unsigned short* __restrict__ W,   // [N x K] bf16 row-major
                 float* __restrict__ C,                  // [M x N] f32
                 int M, int N, int K) {
    __shared__ unsigned short bsm[2][128 * 32];          // 2 x 8 KB B-tile buffers

    const int tid   = threadIdx.x;
    const int lane  = tid & 31;
    const int wave  = tid >> 5;                          // 0..7
    const int laneM = lane & 15;
    const int hi    = lane >> 4;                         // 0 or 1
    const int mBase = blockIdx.y * 128 + wave * 16;
    const int nBase = blockIdx.x * 128;

    // staging role: each thread owns a contiguous 32B chunk of the 128x32 tile
    //   row sr = tid>>1 (0..127), element offset sc = (tid&1)*16 (0 or 16)
    const int sr = tid >> 1;
    const int sc = (tid & 1) * 16;
    const unsigned short* wStage = W + (size_t)(nBase + sr) * K + sc;
    const unsigned ldsStage0 =
        (unsigned)(unsigned long long)(&bsm[0][sr * 32 + sc]); // LDS byte addr

    v8f acc[8];
    #pragma unroll
    for (int n = 0; n < 8; ++n) acc[n] = (v8f)0.0f;

    // A fragment per-lane K offsets: lanes 0-15 -> {0,16}, lanes 16-31 -> {8,24}
    const unsigned short* aRow = A + (size_t)(mBase + laneM) * K + hi * 8;
    // per-lane base of this wave's B fragments inside the LDS tile
    const int bOff = laneM * 32 + hi * 8;

    // ---- prologue: async-stage K-step 0, preload A frag 0 ----
    asyncCopy16B(ldsStage0,      wStage);
    asyncCopy16B(ldsStage0 + 16, wStage + 8);
    Frag16 aCur;
    aCur.u[0] = load16B(aRow);
    aCur.u[1] = load16B(aRow + 16);
    waitAsync0();
    __syncthreads();

    const int steps = K >> 5;                            // >= 1
    // ---- steady state: steps-1 iterations, no guards ----
    #pragma unroll 2
    for (int s = 0; s < steps - 1; ++s) {
        const int cur = s & 1;
        const int kNext = (s + 1) << 5;

        // async-stage NEXT B tile into back buffer + next A frag to registers;
        // both overlap the 8 WMMAs below
        const unsigned dst = ldsStage0 + (1 - cur) * (128 * 32 * 2);
        asyncCopy16B(dst,      wStage + kNext);
        asyncCopy16B(dst + 16, wStage + kNext + 8);
        Frag16 aNext;
        aNext.u[0] = load16B(aRow + kNext);
        aNext.u[1] = load16B(aRow + kNext + 16);

        const unsigned short* bBase = &bsm[cur][bOff];
        #pragma unroll
        for (int n = 0; n < 8; ++n) {
            Frag16 b;
            b.u[0] = *reinterpret_cast<const u32x4*>(bBase + n * 16 * 32);
            b.u[1] = *reinterpret_cast<const u32x4*>(bBase + n * 16 * 32 + 16);
            acc[n] = __builtin_amdgcn_wmma_f32_16x16x32_bf16(
                false, aCur.v, false, b.v, (short)0, acc[n], false, false);
        }

        aCur = aNext;
        waitAsync0();          // async writes of back buffer complete...
        __syncthreads();       // ...then published to all waves
    }

    // ---- epilogue: last K-step, buffer already published ----
    {
        const int cur = (steps - 1) & 1;
        const unsigned short* bBase = &bsm[cur][bOff];
        #pragma unroll
        for (int n = 0; n < 8; ++n) {
            Frag16 b;
            b.u[0] = *reinterpret_cast<const u32x4*>(bBase + n * 16 * 32);
            b.u[1] = *reinterpret_cast<const u32x4*>(bBase + n * 16 * 32 + 16);
            acc[n] = __builtin_amdgcn_wmma_f32_16x16x32_bf16(
                false, aCur.v, false, b.v, (short)0, acc[n], false, false);
        }
    }

    // D layout: VGPR r -> row (hi*8 + r), col = laneM
    #pragma unroll
    for (int n = 0; n < 8; ++n) {
        #pragma unroll
        for (int r = 0; r < 8; ++r) {
            size_t row = (size_t)(mBase + hi * 8 + r);
            C[row * N + nBase + n * 16 + laneM] = acc[n][r];
        }
    }
}

// ---------------------------------------------------------------------------
// 4) Recurrent scan, single workgroup (32 waves), h resident in LDS.
//    h_new = tanh(xin_t + h @ w_hh^T);  B=16 (one M-tile), H=1024 (64 N-tiles)
//    Each wave owns 2 N-tiles; w_hh fragments double-buffered in registers
//    (last k-step peeled -- no guards in the hot loop).
// ---------------------------------------------------------------------------
#define RNN_H 1024
#define RNN_B 16

__global__ void __launch_bounds__(1024)
k_rnn_scan(const unsigned short* __restrict__ whh,  // [H x H] bf16
           const float* __restrict__ xin,           // [B*S x H] f32, row = b*S + t
           unsigned short* __restrict__ hs,         // [B*S x H] bf16
           int S) {
    __shared__ unsigned short hbuf[RNN_B * RNN_H];   // 32 KB

    const int tid   = threadIdx.x;
    const int lane  = tid & 31;
    const int wave  = tid >> 5;                      // 0..31
    const int laneM = lane & 15;
    const int hi    = lane >> 4;

    // zero-init h
    #pragma unroll
    for (int i = 0; i < (RNN_B * RNN_H) / 1024; ++i)
        hbuf[tid + i * 1024] = 0;
    __syncthreads();

    const int n0 = (wave * 2) * 16;                  // first N-tile base col
    const int n1 = (wave * 2 + 1) * 16;
    const unsigned short* w0 = whh + (size_t)(n0 + laneM) * RNN_H + hi * 16;
    const unsigned short* w1 = whh + (size_t)(n1 + laneM) * RNN_H + hi * 16;
    const unsigned short* aRow = hbuf + laneM * RNN_H + hi * 16;

    for (int t = 0; t < S; ++t) {
        v8f acc0 = (v8f)0.0f, acc1 = (v8f)0.0f;

        // weight double-buffer: preload k=0 fragments
        Frag16 c0, c1;
        c0.u[0] = load16B(w0);  c0.u[1] = load16B(w0 + 8);
        c1.u[0] = load16B(w1);  c1.u[1] = load16B(w1 + 8);

        #pragma unroll 2
        for (int k0 = 0; k0 < RNN_H - 32; k0 += 32) {
            // issue next-k weight loads first (long latency)
            Frag16 n0f, n1f;
            n0f.u[0] = load16B(w0 + k0 + 32);  n0f.u[1] = load16B(w0 + k0 + 40);
            n1f.u[0] = load16B(w1 + k0 + 32);  n1f.u[1] = load16B(w1 + k0 + 40);
            Frag16 a;                                 // from LDS (ds_load_b128)
            a.u[0] = *reinterpret_cast<const u32x4*>(aRow + k0);
            a.u[1] = *reinterpret_cast<const u32x4*>(aRow + k0 + 8);
            acc0 = __builtin_amdgcn_wmma_f32_16x16x32_bf16(
                false, a.v, false, c0.v, (short)0, acc0, false, false);
            acc1 = __builtin_amdgcn_wmma_f32_16x16x32_bf16(
                false, a.v, false, c1.v, (short)0, acc1, false, false);
            c0 = n0f;  c1 = n1f;
        }
        {   // peeled last k-step
            const int k0 = RNN_H - 32;
            Frag16 a;
            a.u[0] = *reinterpret_cast<const u32x4*>(aRow + k0);
            a.u[1] = *reinterpret_cast<const u32x4*>(aRow + k0 + 8);
            acc0 = __builtin_amdgcn_wmma_f32_16x16x32_bf16(
                false, a.v, false, c0.v, (short)0, acc0, false, false);
            acc1 = __builtin_amdgcn_wmma_f32_16x16x32_bf16(
                false, a.v, false, c1.v, (short)0, acc1, false, false);
        }
        __syncthreads();   // all reads of hbuf done before overwriting

        #pragma unroll
        for (int r = 0; r < 8; ++r) {
            int m = hi * 8 + r;                       // batch row 0..15
            size_t xrow = ((size_t)m * S + t) * RNN_H;
            {
                int col = n0 + laneM;
                float v = tanhf(acc0[r] + xin[xrow + col]);
                unsigned short bv = f32_to_bf16_rne(v);
                hbuf[m * RNN_H + col] = bv;
                hs[xrow + col] = bv;
            }
            {
                int col = n1 + laneM;
                float v = tanhf(acc1[r] + xin[xrow + col]);
                unsigned short bv = f32_to_bf16_rne(v);
                hbuf[m * RNN_H + col] = bv;
                hs[xrow + col] = bv;
            }
        }
        __syncthreads();   // h_new visible before next step's reads
    }
}

// ---------------------------------------------------------------------------
// Launch
// ---------------------------------------------------------------------------
extern "C" void kernel_launch(void* const* d_in, const int* in_sizes, int n_in,
                              void* d_out, int out_size, void* d_ws, size_t ws_size,
                              hipStream_t stream) {
    const int B = 16, S = 256, H = 1024, V = 32000;
    const int BS = B * S;                  // 4096

    const int*   ids   = (const int*)d_in[0];
    const float* emb   = (const float*)d_in[1];
    const float* w_hx  = (const float*)d_in[2];
    const float* w_hh  = (const float*)d_in[3];
    const float* w_yh  = (const float*)d_in[4];
    float* logits      = (float*)d_out;

    // workspace partition (bytes)
    char* ws = (char*)d_ws;
    unsigned short* x_bf   = (unsigned short*)(ws);                         // 8 MB
    unsigned short* whx_bf = (unsigned short*)(ws + 8388608);               // 2 MB
    unsigned short* whh_bf = (unsigned short*)(ws + 10485760);              // 2 MB
    unsigned short* wyh_bf = (unsigned short*)(ws + 12582912);              // 65.5 MB
    float*          xin    = (float*)        (ws + 78118912);               // 16.8 MB
    unsigned short* hs_bf  = (unsigned short*)(ws + 94896128);              // 8 MB

    // 1) convert weights to bf16
    k_convert_bf16<<<(H * H) / 256, 256, 0, stream>>>(w_hx, whx_bf, H * H);
    k_convert_bf16<<<(H * H) / 256, 256, 0, stream>>>(w_hh, whh_bf, H * H);
    k_convert_bf16<<<(V * H) / 256, 256, 0, stream>>>(w_yh, wyh_bf, V * H);

    // 2) embedding gather -> bf16
    k_gather_emb<<<(BS * H) / 256, 256, 0, stream>>>(ids, emb, x_bf, H);

    // 3) xin = x @ w_hx^T   (M=4096, N=1024, K=1024)
    k_wmma_gemm_bf16<<<dim3(H / 128, BS / 128), 256, 0, stream>>>(
        x_bf, whx_bf, xin, BS, H, H);

    // 4) recurrent scan (single WGP, h in LDS)
    k_rnn_scan<<<1, 1024, 0, stream>>>(whh_bf, xin, hs_bf, S);

    // 5) logits = hs @ w_yh^T   (M=4096, N=32000, K=1024)
    k_wmma_gemm_bf16<<<dim3(V / 128, BS / 128), 256, 0, stream>>>(
        hs_bf, wyh_bf, logits, BS, V, H);
}
